// EnergyTransformerLayer_15582141350439
// MI455X (gfx1250) — matrix operand order repaired and checked by reading, not compiled, not run on hardware
//
#include <hip/hip_runtime.h>
#include <hip/hip_bf16.h>
#include <math.h>

#define EMBED  1024
#define NHEADS 16
#define HD     64
#define HIDDEN 4096
#define NCTX   2048
#define NTGT   1024
#define ESTEPS 5
#define STEPSZ 0.1f
#define BETA_S 0.125f   /* BETA / sqrt(HEAD_DIM) = 1/8 */

typedef __bf16 v16bf __attribute__((ext_vector_type(16)));
typedef __bf16 v8bf  __attribute__((ext_vector_type(8)));
typedef float  v8f   __attribute__((ext_vector_type(8)));
typedef unsigned int   u32;
typedef unsigned short u16;
typedef unsigned long long u64;
typedef u32 v4u __attribute__((ext_vector_type(4)));
typedef int v4i __attribute__((ext_vector_type(4)));
typedef int v8i __attribute__((ext_vector_type(8)));

#if __has_builtin(__builtin_amdgcn_tensor_load_to_lds)
#define HAS_TDM 1
#endif

__device__ __forceinline__ u16 f2bf(float f) {
  u32 u = __float_as_uint(f);
  u32 r = u + 0x7FFFu + ((u >> 16) & 1u);   // round-to-nearest-even
  return (u16)(r >> 16);
}

__device__ __forceinline__ v16bf cat8(v8bf lo, v8bf hi) {
  return __builtin_shufflevector(lo, hi, 0,1,2,3,4,5,6,7,8,9,10,11,12,13,14,15);
}

__device__ __forceinline__ v8f wmma_bf16(v16bf a, v16bf b, v8f c) {
  // (neg_a, A, neg_b, B, c_mod, C, reuse_a, reuse_b) -> v_wmma_f32_16x16x32_bf16
  return __builtin_amdgcn_wmma_f32_16x16x32_bf16(false, a, false, b, (short)0, c,
                                                 false, false);
}

// async copy global(16B) -> LDS, tracked by ASYNCcnt (bypasses VGPRs)
__device__ __forceinline__ void async_b128(u32 lds_addr, const u16* gaddr) {
  asm volatile("global_load_async_to_lds_b128 %0, %1, off"
               :: "v"(lds_addr), "v"(gaddr) : "memory");
}
__device__ __forceinline__ void async_b128_off16(u32 lds_addr, const u16* gaddr) {
  asm volatile("global_load_async_to_lds_b128 %0, %1, off offset:16"
               :: "v"(lds_addr), "v"(gaddr) : "memory");
}

// ---------------------------------------------------------------------------
// Elementwise conversion kernels
// ---------------------------------------------------------------------------
__global__ void cvt_f32_bf16(const float* __restrict__ in, u16* __restrict__ out, int n) {
  int i = blockIdx.x * blockDim.x + threadIdx.x;
  if (i < n) out[i] = f2bf(in[i]);
}

__global__ void tanh_cvt(const float* __restrict__ in, const float* __restrict__ alpha,
                         u16* __restrict__ out, int n) {
  int i = blockIdx.x * blockDim.x + threadIdx.x;
  if (i < n) out[i] = f2bf(tanhf(alpha[0] * in[i]));
}

__global__ void gelu_cvt(const float* __restrict__ in, u16* __restrict__ out, int n) {
  int i = blockIdx.x * blockDim.x + threadIdx.x;
  if (i < n) {
    float x = in[i];
    float t = tanhf(0.7978845608028654f * (x + 0.044715f * x * x * x));
    out[i] = f2bf(0.5f * x * (1.f + t));
  }
}

// ---------------------------------------------------------------------------
// Tiled WMMA GEMM:  C[M,N] = scale * A[M,K] * B[N,K]^T (+ Cin)
// Block = 256 threads (8 waves), tile 64x128, BK=32, async-to-LDS double buffer.
// Wave w: rows (w&3)*16, cols (w>>2)*64 -> 4 WMMA per k-chunk per wave.
// ---------------------------------------------------------------------------
template<bool ADD_C, bool OUT_BF16>
__global__ void __launch_bounds__(256)
gemm_nt_wmma(const u16* __restrict__ A, int lda,
             const u16* __restrict__ B, int ldb,
             void* __restrict__ Cout, int ldc,
             const float* __restrict__ Cin, int ldcin,
             int Kd, float scale) {
  __shared__ __align__(16) u16 As[2][64][40];    // [m][k] pad -> 16B frag runs
  __shared__ __align__(16) u16 Bs[2][128][40];   // [n][k]

  const int m0 = blockIdx.y * 64, n0 = blockIdx.x * 128;
  const int t = threadIdx.x;
  const int lane = t & 31, w = t >> 5;
  const int wm = w & 3, wn = w >> 2;
  const int nn = lane & 15, hs = lane >> 4;
  const int ar = t >> 2, ac = (t & 3) * 8;   // A loader: 64 rows x 32k, 16B/thread
  const int br = t >> 1, bc = (t & 1) * 16;  // B loader: 128 rows x 32k, 32B/thread

  auto issue = [&](int k0, int buf) {
    async_b128((u32)(size_t)&As[buf][ar][ac], A + (size_t)(m0 + ar) * lda + k0 + ac);
    const u16* gb = B + (size_t)(n0 + br) * ldb + k0 + bc;
    u32 lb = (u32)(size_t)&Bs[buf][br][bc];
    async_b128(lb, gb);
    async_b128_off16(lb, gb);
  };

  v8f acc[4] = {};
  issue(0, 0);
  int cur = 0;
  for (int k0 = 0; k0 < Kd; k0 += 32) {
    if (k0 + 32 < Kd) {
      issue(k0 + 32, cur ^ 1);                       // prefetch next tile
      asm volatile("s_wait_asynccnt 0x3" ::: "memory");  // older 3 (in-order) done
    } else {
      asm volatile("s_wait_asynccnt 0x0" ::: "memory");
    }
    __syncthreads();

    // A fragment: lane m = wm*16+nn; runs k = 8*hs..+7 and 16+8*hs..+7
    const int am = wm * 16 + nn;
    v8bf alo = *(const v8bf*)&As[cur][am][8 * hs];
    v8bf ahi = *(const v8bf*)&As[cur][am][16 + 8 * hs];
    v16bf af = cat8(alo, ahi);
#pragma unroll
    for (int j = 0; j < 4; ++j) {
      const int bn = wn * 64 + j * 16 + nn;          // B frag: runs k = 8*hs..+7 and 16+8*hs..+7
      v8bf blo = *(const v8bf*)&Bs[cur][bn][8 * hs];
      v8bf bhi = *(const v8bf*)&Bs[cur][bn][16 + 8 * hs];
      acc[j] = wmma_bf16(af, cat8(blo, bhi), acc[j]);
    }
    __syncthreads();
    cur ^= 1;
  }

  // Epilogue: C layout rows m = r + 8*hs, col = lane&15
#pragma unroll
  for (int j = 0; j < 4; ++j)
#pragma unroll
    for (int r = 0; r < 8; ++r) {
      int row = m0 + wm * 16 + r + 8 * hs;
      int col = n0 + wn * 64 + j * 16 + nn;
      float v = scale * acc[j][r] + (ADD_C ? Cin[(size_t)row * ldcin + col] : 0.f);
      if (OUT_BF16) ((u16*)Cout)[(size_t)row * ldc + col] = f2bf(v);
      else          ((float*)Cout)[(size_t)row * ldc + col] = v;
    }
}

// ---------------------------------------------------------------------------
// Fused 5-step flash energy-attention.
// Grid: (NTGT/64, NHEADS). Block: 128 threads = 4 waves, 16 q-rows per wave.
// K chunks stream through LDS via the Tensor Data Mover (descriptor does the
// row padding: pad_interval=4 -> every 32 DWORDs, pad_amount=3 -> 4 DWORDs,
// matching Kt[32][72]); falls back to async-to-LDS loads.
// ---------------------------------------------------------------------------
__global__ void __launch_bounds__(128)
energy_attn(const float* __restrict__ q0, const u16* __restrict__ Kb,
            u16* __restrict__ qf) {
  __shared__ __align__(16) u16 Kt [32][72];   // K chunk [ctx][z]
  __shared__ __align__(16) u16 KtT[64][40];   // transposed [z][ctx] for P*K B-frags
  __shared__ __align__(16) u16 Qs[4][16][72]; // per-wave q bf16 relay (C->A layout)
  __shared__ __align__(16) u16 Ps[4][16][40]; // per-wave P bf16 relay

  const int h  = blockIdx.y;
  const int qb = blockIdx.x * 64;
  const int t = threadIdx.x, lane = t & 31, w = t >> 5;
  const int nn = lane & 15, hs = lane >> 4;

  // q rows in registers, C layout: qc[zt][r] = q[qb+w*16+r+8*hs][h*64+16*zt+nn]
  v8f qc[4];
#pragma unroll
  for (int zt = 0; zt < 4; ++zt)
#pragma unroll
    for (int r = 0; r < 8; ++r)
      qc[zt][r] = q0[(size_t)(qb + w * 16 + r + 8 * hs) * EMBED + h * HD + zt * 16 + nn];

  const int lr = t >> 2;         // chunk worker: ctx row 0..31
  const int lz = (t & 3) * 16;   // z base (16 elems = 32B)

  for (int step = 0; step < ESTEPS; ++step) {
    // stage q as bf16, rebuild A-fragments (same-wave DS ops are in-order)
#pragma unroll
    for (int zt = 0; zt < 4; ++zt)
#pragma unroll
      for (int r = 0; r < 8; ++r)
        Qs[w][r + 8 * hs][zt * 16 + nn] = f2bf(qc[zt][r]);
    v16bf qa[2];
#pragma unroll
    for (int zc = 0; zc < 2; ++zc) {
      v8bf lo = *(const v8bf*)&Qs[w][nn][zc * 32 + 8 * hs];
      v8bf hi = *(const v8bf*)&Qs[w][nn][zc * 32 + 16 + 8 * hs];
      qa[zc] = cat8(lo, hi);
    }

    float mi[8], li[8];
    v8f o[4];
#pragma unroll
    for (int r = 0; r < 8; ++r) { mi[r] = -1e30f; li[r] = 0.f; }
#pragma unroll
    for (int zt = 0; zt < 4; ++zt) o[zt] = (v8f){};

    for (int k0 = 0; k0 < NCTX; k0 += 32) {
      __syncthreads();
#ifdef HAS_TDM
      if (w == 0) {   // one TDM descriptor loads the whole 32x64 K tile
        u64 ga = (u64)(size_t)(Kb + (size_t)k0 * EMBED + h * HD);
        v4u g0;
        g0.x = 1u;                                   // count=1, no gather
        g0.y = (u32)(size_t)&Kt[0][0];               // lds_addr
        g0.z = (u32)ga;                              // global_addr[31:0]
        g0.w = (u32)((ga >> 32) & 0x01FFFFFFu) | (2u << 30);  // addr[56:32]|type=2
        v8i g1;
        g1[0] = (int)((1u << 16) |                   // data_size = 2B
                      (1u << 20) |                   // pad_enable
                      (4u << 22) |                   // pad_interval: 32 DWORDs
                      (3u << 25));                   // pad_amount: 4 DWORDs
        g1[1] = (int)(((u32)EMBED & 0xFFFFu) << 16);            // tensor_dim0 lo
        g1[2] = (int)(((u32)EMBED >> 16) | (((u32)NCTX & 0xFFFFu) << 16));
        g1[3] = (int)(((u32)NCTX >> 16) | (64u << 16));         // tile_dim0=64
        g1[4] = 32;                                  // tile_dim1
        g1[5] = EMBED;                               // tensor_dim0_stride lo
        g1[6] = 0; g1[7] = 0;
        v4i z4 = {0, 0, 0, 0};
#if __clang_major__ >= 23
        v8i z8 = {0, 0, 0, 0, 0, 0, 0, 0};
        __builtin_amdgcn_tensor_load_to_lds(g0, g1, z4, z4, z8, 0);
#else
        __builtin_amdgcn_tensor_load_to_lds(g0, g1, z4, z4, 0);
#endif
        __builtin_amdgcn_s_wait_tensorcnt(0);
      }
#else
      {
        const u16* g = Kb + (size_t)(k0 + lr) * EMBED + h * HD + lz;
        u32 la = (u32)(size_t)&Kt[lr][lz];
        async_b128(la, g);
        async_b128_off16(la, g);
        asm volatile("s_wait_asynccnt 0x0" ::: "memory");
      }
#endif
      __syncthreads();
      { // build transposed copy for the P*K B-fragments
        uint4 va = *(const uint4*)&Kt[lr][lz];
        uint4 vb = *(const uint4*)&Kt[lr][lz + 8];
        u32 wds[8] = {va.x, va.y, va.z, va.w, vb.x, vb.y, vb.z, vb.w};
#pragma unroll
        for (int i = 0; i < 8; ++i) {
          KtT[lz + 2 * i][lr]     = (u16)(wds[i] & 0xffffu);
          KtT[lz + 2 * i + 1][lr] = (u16)(wds[i] >> 16);
        }
      }
      __syncthreads();

      // scores S[16 q x 32 ctx] = q(16x64) * Kchunk^T : 4 WMMA
      v8f s[2];
#pragma unroll
      for (int j = 0; j < 2; ++j) {
        v8f c = {};
#pragma unroll
        for (int zc = 0; zc < 2; ++zc) {
          v8bf lo = *(const v8bf*)&Kt[j * 16 + nn][zc * 32 + 8 * hs];
          v8bf hi = *(const v8bf*)&Kt[j * 16 + nn][zc * 32 + 16 + 8 * hs];
          c = wmma_bf16(qa[zc], cat8(lo, hi), c);
        }
        s[j] = c;
      }

      // online softmax (row = q index; reduce across 16 lanes of half-wave)
#pragma unroll
      for (int r = 0; r < 8; ++r) {
        float v = fmaxf(s[0][r], s[1][r]) * BETA_S;
#pragma unroll
        for (int msk = 1; msk < 16; msk <<= 1)
          v = fmaxf(v, __shfl_xor(v, msk, 32));
        float mn = fmaxf(mi[r], v);
        float al = __expf(mi[r] - mn);
        float p0 = __expf(BETA_S * s[0][r] - mn);
        float p1 = __expf(BETA_S * s[1][r] - mn);
        float rs = p0 + p1;
#pragma unroll
        for (int msk = 1; msk < 16; msk <<= 1)
          rs += __shfl_xor(rs, msk, 32);
        li[r] = li[r] * al + rs;
        mi[r] = mn;
#pragma unroll
        for (int zt = 0; zt < 4; ++zt) o[zt][r] *= al;
        Ps[w][r + 8 * hs][nn]      = f2bf(p0);
        Ps[w][r + 8 * hs][16 + nn] = f2bf(p1);
      }

      // O[16x64] += P[16x32] * Kchunk[32x64] : 4 WMMA
      v8bf plo = *(const v8bf*)&Ps[w][nn][8 * hs];
      v8bf phi = *(const v8bf*)&Ps[w][nn][16 + 8 * hs];
      v16bf pf = cat8(plo, phi);
#pragma unroll
      for (int zt = 0; zt < 4; ++zt) {
        v8bf blo = *(const v8bf*)&KtT[zt * 16 + nn][8 * hs];
        v8bf bhi = *(const v8bf*)&KtT[zt * 16 + nn][16 + 8 * hs];
        o[zt] = wmma_bf16(pf, cat8(blo, bhi), o[zt]);
      }
    } // ctx chunks

    // q -= STEP * grad  (grad = -probs@K)  =>  q += STEP * O / l
#pragma unroll
    for (int zt = 0; zt < 4; ++zt)
#pragma unroll
      for (int r = 0; r < 8; ++r)
        qc[zt][r] += STEPSZ * o[zt][r] / li[r];
  } // energy steps

  // emit final q directly as bf16 (input to the Wo GEMM)
#pragma unroll
  for (int zt = 0; zt < 4; ++zt)
#pragma unroll
    for (int r = 0; r < 8; ++r)
      qf[(size_t)(qb + w * 16 + r + 8 * hs) * EMBED + h * HD + zt * 16 + nn] =
          f2bf(qc[zt][r]);
}

// ---------------------------------------------------------------------------
extern "C" void kernel_launch(void* const* d_in, const int* in_sizes, int n_in,
                              void* d_out, int out_size, void* d_ws, size_t ws_size,
                              hipStream_t stream) {
  (void)in_sizes; (void)n_in; (void)out_size; (void)ws_size;
  const float* ctx = (const float*)d_in[0];
  const float* tgt = (const float*)d_in[1];
  const float* Wq  = (const float*)d_in[2];
  const float* Wk  = (const float*)d_in[3];
  const float* Wo  = (const float*)d_in[4];
  const float* W1  = (const float*)d_in[5];
  const float* W2  = (const float*)d_in[6];
  const float* a1  = (const float*)d_in[7];
  const float* a2  = (const float*)d_in[8];
  float* out = (float*)d_out;

  char* p = (char*)d_ws;
  auto alloc = [&](size_t bytes) {
    char* r = p; p += (bytes + 255) & ~(size_t)255; return r;
  };
  u16*   ctx_bf = (u16*)alloc((size_t)NCTX * EMBED * 2);
  u16*   tn_bf  = (u16*)alloc((size_t)NTGT * EMBED * 2);
  u16*   Wq_bf  = (u16*)alloc((size_t)EMBED * EMBED * 2);
  u16*   Wk_bf  = (u16*)alloc((size_t)EMBED * EMBED * 2);
  u16*   Wo_bf  = (u16*)alloc((size_t)EMBED * EMBED * 2);
  u16*   W1_bf  = (u16*)alloc((size_t)HIDDEN * EMBED * 2);
  u16*   W2_bf  = (u16*)alloc((size_t)EMBED * HIDDEN * 2);
  u16*   K_bf   = (u16*)alloc((size_t)NCTX * EMBED * 2);
  float* q0     = (float*)alloc((size_t)NTGT * EMBED * 4);
  u16*   qf_bf  = (u16*)alloc((size_t)NTGT * EMBED * 2);
  float* xres   = (float*)alloc((size_t)NTGT * EMBED * 4);
  u16*   t2_bf  = (u16*)alloc((size_t)NTGT * EMBED * 2);
  float* hbuf   = (float*)alloc((size_t)NTGT * HIDDEN * 4);
  u16*   hg_bf  = (u16*)alloc((size_t)NTGT * HIDDEN * 2);

  auto blks = [](int n) { return (n + 255) / 256; };

  // 0) bf16 staging of activations and weights
  cvt_f32_bf16<<<blks(NCTX * EMBED), 256, 0, stream>>>(ctx, ctx_bf, NCTX * EMBED);
  tanh_cvt    <<<blks(NTGT * EMBED), 256, 0, stream>>>(tgt, a1, tn_bf, NTGT * EMBED);
  cvt_f32_bf16<<<blks(EMBED * EMBED), 256, 0, stream>>>(Wq, Wq_bf, EMBED * EMBED);
  cvt_f32_bf16<<<blks(EMBED * EMBED), 256, 0, stream>>>(Wk, Wk_bf, EMBED * EMBED);
  cvt_f32_bf16<<<blks(EMBED * EMBED), 256, 0, stream>>>(Wo, Wo_bf, EMBED * EMBED);
  cvt_f32_bf16<<<blks(HIDDEN * EMBED), 256, 0, stream>>>(W1, W1_bf, HIDDEN * EMBED);
  cvt_f32_bf16<<<blks(EMBED * HIDDEN), 256, 0, stream>>>(W2, W2_bf, EMBED * HIDDEN);

  // 1) K = context @ WkAll^T  -> bf16 [2048,1024] (col = h*64+z)
  gemm_nt_wmma<false, true><<<dim3(EMBED / 128, NCTX / 64), 256, 0, stream>>>(
      ctx_bf, EMBED, Wk_bf, EMBED, K_bf, EMBED, nullptr, 0, EMBED, 1.f);

  // 2) q0 = tanh(a1*target) @ WqAll^T -> f32 [1024,1024]
  gemm_nt_wmma<false, false><<<dim3(EMBED / 128, NTGT / 64), 256, 0, stream>>>(
      tn_bf, EMBED, Wq_bf, EMBED, q0, EMBED, nullptr, 0, EMBED, 1.f);

  // 3) fused 5-step energy descent (flash attention with V=K) -> q bf16
  energy_attn<<<dim3(NTGT / 64, NHEADS), 128, 0, stream>>>(q0, K_bf, qf_bf);

  // 4) x = target + q @ Wo^T
  gemm_nt_wmma<true, false><<<dim3(EMBED / 128, NTGT / 64), 256, 0, stream>>>(
      qf_bf, EMBED, Wo_bf, EMBED, xres, EMBED, tgt, EMBED, EMBED, 1.f);

  // 5) FFN
  tanh_cvt<<<blks(NTGT * EMBED), 256, 0, stream>>>(xres, a2, t2_bf, NTGT * EMBED);
  gemm_nt_wmma<false, false><<<dim3(HIDDEN / 128, NTGT / 64), 256, 0, stream>>>(
      t2_bf, EMBED, W1_bf, EMBED, hbuf, HIDDEN, nullptr, 0, EMBED, 1.f);
  gelu_cvt<<<blks(NTGT * HIDDEN), 256, 0, stream>>>(hbuf, hg_bf, NTGT * HIDDEN);
  gemm_nt_wmma<true, false><<<dim3(EMBED / 128, NTGT / 64), 256, 0, stream>>>(
      hg_bf, HIDDEN, W2_bf, HIDDEN, out, EMBED, xres, EMBED, HIDDEN, 1.f);
}
